// Encoder_28381143892815
// MI455X (gfx1250) — compile-verified, matrix operand order checked by reading, not run on hardware
//
#include <hip/hip_runtime.h>

typedef _Float16 f16;
typedef __attribute__((ext_vector_type(16))) _Float16 v16h;
typedef __attribute__((ext_vector_type(8)))  float    v8f;
typedef __attribute__((ext_vector_type(8)))  unsigned int v8u;

#define S_LEN   512
#define BATCH   32
#define FDIM    256
#define HDIM    256
#define NNODES  20000
#define NEDGES  320000

// ---------------------------------------------------------------------------
// WMMA helpers (CDNA5 16x16x32 f16 -> f32)
// ---------------------------------------------------------------------------
__device__ __forceinline__ v16h load_frag16(const f16* base, int lane) {
  // 16-bit A/B fragment: lane = row (mod 16), hi half of wave offsets K by 8.
  // VGPR v<4 : K = 2v + 8*hi ; v>=4 : K = 16 + 2(v-4) + 8*hi. Pairs packed.
  const unsigned int* pu = (const unsigned int*)base;
  int hi = (lane >> 4) & 1;
  v8u u;
#pragma unroll
  for (int v = 0; v < 8; ++v) {
    int kofs = 2 * v + 8 * hi + ((v >= 4) ? 8 : 0);
    u[v] = pu[kofs >> 1];
  }
  return __builtin_bit_cast(v16h, u);
}

__device__ __forceinline__ v8f wmma32(v16h a, v16h b, v8f c) {
  return __builtin_amdgcn_wmma_f32_16x16x32_f16(false, a, false, b,
                                                (short)0, c, false, false);
}

__device__ __forceinline__ float sigmoidf_(float x) {
  return 1.0f / (1.0f + __expf(-x));
}

// ---------------------------------------------------------------------------
// Generic WMMA GEMM:  C[M,N](f32) = A[M,K](f16) @ W[N,K](f16)^T (+ bias[N])
// grid.x = M/32, grid.y = N/256, block = 128 (4 waves).
// Each wave: 2 M-tiles x 4 N-tiles (32x64), register double-buffered K loop.
// ---------------------------------------------------------------------------
__global__ __launch_bounds__(128) void wmma_gemm_kernel(
    const f16* __restrict__ A, int lda,
    const f16* __restrict__ W, int ldw,
    const float* __restrict__ bias,
    float* __restrict__ C, int ldc, int K) {
  int lane = threadIdx.x & 31;
  int wave = threadIdx.x >> 5;
  int m0 = blockIdx.x * 32;
  int n0 = blockIdx.y * 256 + wave * 64;

  const f16* a0 = A + (size_t)(m0 +      (lane & 15)) * lda;
  const f16* a1 = A + (size_t)(m0 + 16 + (lane & 15)) * lda;
  const f16* b0 = W + (size_t)(n0 +  0 + (lane & 15)) * ldw;
  const f16* b1 = W + (size_t)(n0 + 16 + (lane & 15)) * ldw;
  const f16* b2 = W + (size_t)(n0 + 32 + (lane & 15)) * ldw;
  const f16* b3 = W + (size_t)(n0 + 48 + (lane & 15)) * ldw;

  v8f acc0 = {}, acc1 = {}, acc2 = {}, acc3 = {};
  v8f acc4 = {}, acc5 = {}, acc6 = {}, acc7 = {};

  // prologue: fragments for K-step 0
  v16h ca0 = load_frag16(a0, lane);
  v16h ca1 = load_frag16(a1, lane);
  v16h cb0 = load_frag16(b0, lane);
  v16h cb1 = load_frag16(b1, lane);
  v16h cb2 = load_frag16(b2, lane);
  v16h cb3 = load_frag16(b3, lane);

  for (int kk = 32; kk < K; kk += 32) {
    // prefetch the streamed A operand a couple of K-tiles ahead
    __builtin_prefetch(a0 + kk + 128, 0, 1);
    __builtin_prefetch(a1 + kk + 128, 0, 1);
    // double-buffer: issue next K-step loads before this step's WMMAs
    v16h na0 = load_frag16(a0 + kk, lane);
    v16h na1 = load_frag16(a1 + kk, lane);
    v16h nb0 = load_frag16(b0 + kk, lane);
    v16h nb1 = load_frag16(b1 + kk, lane);
    v16h nb2 = load_frag16(b2 + kk, lane);
    v16h nb3 = load_frag16(b3 + kk, lane);

    acc0 = wmma32(ca0, cb0, acc0);
    acc1 = wmma32(ca0, cb1, acc1);
    acc2 = wmma32(ca0, cb2, acc2);
    acc3 = wmma32(ca0, cb3, acc3);
    acc4 = wmma32(ca1, cb0, acc4);
    acc5 = wmma32(ca1, cb1, acc5);
    acc6 = wmma32(ca1, cb2, acc6);
    acc7 = wmma32(ca1, cb3, acc7);

    ca0 = na0; ca1 = na1;
    cb0 = nb0; cb1 = nb1; cb2 = nb2; cb3 = nb3;
  }
  // epilogue
  acc0 = wmma32(ca0, cb0, acc0);
  acc1 = wmma32(ca0, cb1, acc1);
  acc2 = wmma32(ca0, cb2, acc2);
  acc3 = wmma32(ca0, cb3, acc3);
  acc4 = wmma32(ca1, cb0, acc4);
  acc5 = wmma32(ca1, cb1, acc5);
  acc6 = wmma32(ca1, cb2, acc6);
  acc7 = wmma32(ca1, cb3, acc7);

  v8f accs[8] = {acc0, acc1, acc2, acc3, acc4, acc5, acc6, acc7};
#pragma unroll
  for (int mt = 0; mt < 2; ++mt) {
    int mbase = m0 + mt * 16 + ((lane >> 4) << 3);
#pragma unroll
    for (int t = 0; t < 4; ++t) {
      int n = n0 + t * 16 + (lane & 15);
      float bv = bias ? bias[n] : 0.0f;
#pragma unroll
      for (int v = 0; v < 8; ++v)
        C[(size_t)(mbase + v) * ldc + n] = accs[mt * 4 + t][v] + bv;
    }
  }
}

// ---------------------------------------------------------------------------
// Persistent recurrent GRU kernel. One block per direction (grid = 2).
// h kept in LDS (f32 + f16 mirror). Per step: gh = h @ whh^T via WMMA
// (96 tiles over 32 waves), then fused gate math.
// ---------------------------------------------------------------------------
__global__ __launch_bounds__(1024) void gru_seq_kernel(
    const float* __restrict__ GI,      // [2][S*B][768]  (x@wih^T + bih)
    const f16*   __restrict__ WHH,     // [2][768][256]
    const float* __restrict__ BHH,     // [2][768]
    f16*         __restrict__ Y16,     // [S][B][512]   (fwd cols 0..255, bwd 256..511)
    float*       __restrict__ hidden,  // [4][B][256]
    int layer) {
  int dir = blockIdx.x;
  const float* gi_base = GI + (size_t)dir * S_LEN * BATCH * 768;
  const f16*   whh     = WHH + (size_t)dir * 768 * 256;
  const float* bhh     = BHH + dir * 768;

  __shared__ f16   h16[BATCH * HDIM];     // 16 KB
  __shared__ float hf [BATCH * HDIM];     // 32 KB
  __shared__ float ghs[BATCH * 3 * HDIM]; // 96 KB

  int tid = threadIdx.x;
  int lane = tid & 31;
  int wave = tid >> 5;

  for (int i = tid; i < BATCH * HDIM; i += 1024) {
    hf[i] = 0.0f;
    h16[i] = (f16)0.0f;
  }
  __syncthreads();

  for (int t = 0; t < S_LEN; ++t) {
    int s = dir ? (S_LEN - 1 - t) : t;

    // ---- gh = h @ whh^T : 2x48 tiles of 16x16, 3 tiles per wave ----
#pragma unroll
    for (int i = 0; i < 3; ++i) {
      int tile = wave * 3 + i;
      int mt = tile / 48, nt = tile % 48;
      int m0 = mt * 16, n0 = nt * 16;
      const f16* arow = h16 + (size_t)(m0 + (lane & 15)) * HDIM;
      const f16* brow = whh + (size_t)(n0 + (lane & 15)) * HDIM;
      v8f acc = {};
#pragma unroll
      for (int kk = 0; kk < HDIM; kk += 32) {
        v16h a = load_frag16(arow + kk, lane);
        v16h b = load_frag16(brow + kk, lane);
        acc = wmma32(a, b, acc);
      }
      int n = n0 + (lane & 15);
      int mbase = m0 + ((lane >> 4) << 3);
#pragma unroll
      for (int v = 0; v < 8; ++v)
        ghs[(mbase + v) * 768 + n] = acc[v];
    }
    __syncthreads();

    // ---- fused gates: 8192 elements over 1024 threads ----
#pragma unroll
    for (int u = 0; u < 8; ++u) {
      int lin = tid + u * 1024;
      int b = lin >> 8, j = lin & 255;
      const float* gi = gi_base + (size_t)(s * BATCH + b) * 768;
      float ghr = ghs[b * 768 + j]       + bhh[j];
      float ghz = ghs[b * 768 + 256 + j] + bhh[256 + j];
      float ghn = ghs[b * 768 + 512 + j] + bhh[512 + j];
      float r  = sigmoidf_(gi[j]       + ghr);
      float z  = sigmoidf_(gi[256 + j] + ghz);
      float nn = tanhf(gi[512 + j] + r * ghn);
      float hnew = (1.0f - z) * nn + z * hf[lin];
      hf[lin]  = hnew;
      h16[lin] = (f16)hnew;
      Y16[(size_t)(s * BATCH + b) * 512 + dir * 256 + j] = (f16)hnew;
    }
    __syncthreads();
  }

  for (int i = tid; i < BATCH * HDIM; i += 1024)
    hidden[(size_t)(2 * layer + dir) * BATCH * HDIM + i] = hf[i];
}

// ---------------------------------------------------------------------------
// Elementwise / data-movement kernels
// ---------------------------------------------------------------------------
__global__ void embed_ref_kernel(const int* __restrict__ input,
                                 const float* __restrict__ emb,
                                 f16* __restrict__ X, int total) {
  int lin = blockIdx.x * 256 + threadIdx.x;
  if (lin >= total) return;
  int f = lin % FDIM;
  int sb = lin / FDIM;
  int b = sb % BATCH;
  int s = sb / BATCH;
  int tok = input[b * S_LEN + s];
  X[lin] = (f16)emb[(size_t)tok * FDIM + f];
}

__global__ void embed_node_kernel(const int* __restrict__ node,
                                  const float* __restrict__ emb,
                                  float* __restrict__ nx,
                                  f16* __restrict__ nx16, int total) {
  int lin = blockIdx.x * 256 + threadIdx.x;
  if (lin >= total) return;
  int h = lin % HDIM;
  int n = lin / HDIM;
  float v = emb[(size_t)node[n] * HDIM + h];
  nx[lin] = v;
  nx16[lin] = (f16)v;
}

__global__ void f32_to_f16_kernel(const float* __restrict__ src,
                                  f16* __restrict__ dst, int n) {
  int i = blockIdx.x * 256 + threadIdx.x;
  if (i < n) dst[i] = (f16)src[i];
}

__global__ void transpose_f32_to_f16_kernel(const float* __restrict__ src,
                                            f16* __restrict__ dst, int n) {
  // src[k][j] (HxH) -> dst[j][k]
  int i = blockIdx.x * 256 + threadIdx.x;
  if (i >= n) return;
  int j = i / HDIM, k = i % HDIM;
  dst[i] = (f16)src[k * HDIM + j];
}

__global__ void zero_f32_kernel(float* __restrict__ p, int n) {
  int i = blockIdx.x * 256 + threadIdx.x;
  if (i < n) p[i] = 0.0f;
}

__global__ void ggc_scatter_kernel(const float* __restrict__ Mlin,
                                   const int* __restrict__ src,
                                   const int* __restrict__ dst,
                                   float* __restrict__ agg, int total) {
  int lin = blockIdx.x * 256 + threadIdx.x;
  if (lin >= total) return;                 // total = E * (H/4)
  int e = lin / (HDIM / 4);
  int c = (lin % (HDIM / 4)) * 4;
  const float4 v = *(const float4*)(Mlin + (size_t)src[e] * HDIM + c);
  float* d = agg + (size_t)dst[e] * HDIM + c;
  atomicAdd(d + 0, v.x);
  atomicAdd(d + 1, v.y);
  atomicAdd(d + 2, v.z);
  atomicAdd(d + 3, v.w);
}

__global__ void ggc_gate_kernel(const float* __restrict__ gi,   // [N][768] incl bih
                                const float* __restrict__ gh,   // [N][768] incl bhh
                                float* __restrict__ nx,
                                f16* __restrict__ nx16, int total) {
  int lin = blockIdx.x * 256 + threadIdx.x;
  if (lin >= total) return;
  int j = lin % HDIM;
  int n = lin / HDIM;
  size_t base = (size_t)n * 768;
  float r  = sigmoidf_(gi[base + j]       + gh[base + j]);
  float z  = sigmoidf_(gi[base + 256 + j] + gh[base + 256 + j]);
  float nn = tanhf(gi[base + 512 + j] + r * gh[base + 512 + j]);
  float h = nx[lin];
  float out = (1.0f - z) * nn + z * h;
  nx[lin] = out;
  nx16[lin] = (f16)out;
}

__global__ void build_cat_kernel(const f16* __restrict__ Y1,     // [S][B][512]
                                 const f16* __restrict__ nx16,   // [N][256]
                                 const int* __restrict__ node_map,
                                 const int* __restrict__ len_ctx,
                                 f16* __restrict__ CAT, int total) {
  int lin = blockIdx.x * 256 + threadIdx.x;
  if (lin >= total) return;                 // total = B*S*768, row = b*S+s
  int j = lin % 768;
  int bs = lin / 768;
  int b = bs / S_LEN;
  int s = bs % S_LEN;
  f16 v;
  if (j < 512) {
    v = Y1[(size_t)(s * BATCH + b) * 512 + j];
  } else {
    int jj = j - 512;
    bool valid = s < len_ctx[b];
    int idx = valid ? node_map[b * S_LEN + s] : 0;
    v = valid ? nx16[(size_t)idx * HDIM + jj] : (f16)0.0f;
  }
  CAT[lin] = v;
}

// ---------------------------------------------------------------------------
// Host launcher
// ---------------------------------------------------------------------------
extern "C" void kernel_launch(void* const* d_in, const int* in_sizes, int n_in,
                              void* d_out, int out_size, void* d_ws, size_t ws_size,
                              hipStream_t stream) {
  const int*   input       = (const int*)d_in[0];
  const int*   node        = (const int*)d_in[1];
  const int*   edge_index  = (const int*)d_in[2];
  const int*   node_map    = (const int*)d_in[3];
  const int*   len_context = (const int*)d_in[4];
  const float* emb_ref     = (const float*)d_in[5];
  const float* emb_node    = (const float*)d_in[6];
  const float* ggc_w       = (const float*)d_in[7];
  const float* ggc_wih     = (const float*)d_in[8];
  const float* ggc_whh     = (const float*)d_in[9];
  const float* ggc_bih     = (const float*)d_in[10];
  const float* ggc_bhh     = (const float*)d_in[11];
  const float* gru_wih0    = (const float*)d_in[12];
  const float* gru_whh0    = (const float*)d_in[13];
  const float* gru_bih0    = (const float*)d_in[14];
  const float* gru_bhh0    = (const float*)d_in[15];
  const float* gru_wih1    = (const float*)d_in[16];
  const float* gru_whh1    = (const float*)d_in[17];
  const float* gru_bih1    = (const float*)d_in[18];
  const float* gru_bhh1    = (const float*)d_in[19];
  const float* W_w         = (const float*)d_in[20];
  const float* W_b         = (const float*)d_in[21];

  float* out_main = (float*)d_out;                                   // [B][S][512]
  float* hidden   = out_main + (size_t)BATCH * S_LEN * 512;          // [4][B][256]

  // ---- workspace bump allocator ----
  size_t off = 0;
  char* ws = (char*)d_ws;
  auto ALLOC = [&](size_t bytes) -> void* {
    void* p = ws + off;
    off += (bytes + 255) & ~(size_t)255;
    return p;
  };
  const size_t SB = (size_t)S_LEN * BATCH;                           // 16384
  f16*   X0    = (f16*)ALLOC(SB * FDIM * 2);
  f16*   Y0    = (f16*)ALLOC(SB * 512 * 2);
  f16*   Y1    = (f16*)ALLOC(SB * 512 * 2);
  float* GI    = (float*)ALLOC(2 * SB * 768 * 4);
  f16*   WIH0  = (f16*)ALLOC((size_t)2 * 768 * 256 * 2);
  f16*   WHH0  = (f16*)ALLOC((size_t)2 * 768 * 256 * 2);
  f16*   WIH1  = (f16*)ALLOC((size_t)2 * 768 * 512 * 2);
  f16*   WHH1  = (f16*)ALLOC((size_t)2 * 768 * 256 * 2);
  f16*   GGCW  = (f16*)ALLOC((size_t)4 * 256 * 256 * 2);
  f16*   GWIH  = (f16*)ALLOC((size_t)768 * 256 * 2);
  f16*   GWHH  = (f16*)ALLOC((size_t)768 * 256 * 2);
  f16*   WW16  = (f16*)ALLOC((size_t)512 * 768 * 2);
  float* NX    = (float*)ALLOC((size_t)NNODES * 256 * 4);
  f16*   NX16  = (f16*)ALLOC((size_t)NNODES * 256 * 2);
  float* MLIN  = (float*)ALLOC((size_t)NNODES * 256 * 4);
  float* AGG   = (float*)ALLOC((size_t)NNODES * 256 * 4);
  f16*   AGG16 = (f16*)ALLOC((size_t)NNODES * 256 * 2);
  float* GGI   = (float*)ALLOC((size_t)NNODES * 768 * 4);
  float* GGH   = (float*)ALLOC((size_t)NNODES * 768 * 4);
  f16*   CAT   = (f16*)ALLOC(SB * 768 * 2);
  (void)ws_size; (void)in_sizes; (void)n_in; (void)out_size;

  auto EW = [](int n) { return dim3((n + 255) / 256); };

  // ---- weight conversion to f16 ----
  f32_to_f16_kernel<<<EW(2 * 768 * 256), 256, 0, stream>>>(gru_wih0, WIH0, 2 * 768 * 256);
  f32_to_f16_kernel<<<EW(2 * 768 * 256), 256, 0, stream>>>(gru_whh0, WHH0, 2 * 768 * 256);
  f32_to_f16_kernel<<<EW(2 * 768 * 512), 256, 0, stream>>>(gru_wih1, WIH1, 2 * 768 * 512);
  f32_to_f16_kernel<<<EW(2 * 768 * 256), 256, 0, stream>>>(gru_whh1, WHH1, 2 * 768 * 256);
  f32_to_f16_kernel<<<EW(768 * 256), 256, 0, stream>>>(ggc_wih, GWIH, 768 * 256);
  f32_to_f16_kernel<<<EW(768 * 256), 256, 0, stream>>>(ggc_whh, GWHH, 768 * 256);
  f32_to_f16_kernel<<<EW(512 * 768), 256, 0, stream>>>(W_w, WW16, 512 * 768);
  for (int i = 0; i < 4; ++i)
    transpose_f32_to_f16_kernel<<<EW(256 * 256), 256, 0, stream>>>(
        ggc_w + (size_t)i * 256 * 256, GGCW + (size_t)i * 256 * 256, 256 * 256);

  // ---- embeddings ----
  embed_ref_kernel<<<EW((int)(SB * FDIM)), 256, 0, stream>>>(input, emb_ref, X0, (int)(SB * FDIM));
  embed_node_kernel<<<EW(NNODES * 256), 256, 0, stream>>>(node, emb_node, NX, NX16, NNODES * 256);

  // ---- GRU layer 0: hoisted input GEMMs + recurrent sweep ----
  for (int dir = 0; dir < 2; ++dir)
    wmma_gemm_kernel<<<dim3((int)(SB / 32), 3), 128, 0, stream>>>(
        X0, FDIM, WIH0 + (size_t)dir * 768 * 256, 256,
        gru_bih0 + dir * 768, GI + (size_t)dir * SB * 768, 768, 256);
  gru_seq_kernel<<<2, 1024, 0, stream>>>(GI, WHH0, gru_bhh0, Y0, hidden, 0);

  // ---- GRU layer 1 ----
  for (int dir = 0; dir < 2; ++dir)
    wmma_gemm_kernel<<<dim3((int)(SB / 32), 3), 128, 0, stream>>>(
        Y0, 512, WIH1 + (size_t)dir * 768 * 512, 512,
        gru_bih1 + dir * 768, GI + (size_t)dir * SB * 768, 768, 512);
  gru_seq_kernel<<<2, 1024, 0, stream>>>(GI, WHH1, gru_bhh1, Y1, hidden, 1);

  // ---- GatedGraphConv: 4 propagation rounds ----
  const int* src = edge_index;
  const int* dst = edge_index + NEDGES;
  for (int i = 0; i < 4; ++i) {
    wmma_gemm_kernel<<<dim3(NNODES / 32, 1), 128, 0, stream>>>(
        NX16, 256, GGCW + (size_t)i * 256 * 256, 256, nullptr, MLIN, 256, 256);
    zero_f32_kernel<<<EW(NNODES * 256), 256, 0, stream>>>(AGG, NNODES * 256);
    ggc_scatter_kernel<<<EW(NEDGES * (HDIM / 4)), 256, 0, stream>>>(MLIN, src, dst, AGG,
                                                                    NEDGES * (HDIM / 4));
    f32_to_f16_kernel<<<EW(NNODES * 256), 256, 0, stream>>>(AGG, AGG16, NNODES * 256);
    wmma_gemm_kernel<<<dim3(NNODES / 32, 3), 128, 0, stream>>>(
        AGG16, 256, GWIH, 256, ggc_bih, GGI, 768, 256);
    wmma_gemm_kernel<<<dim3(NNODES / 32, 3), 128, 0, stream>>>(
        NX16, 256, GWHH, 256, ggc_bhh, GGH, 768, 256);
    ggc_gate_kernel<<<EW(NNODES * 256), 256, 0, stream>>>(GGI, GGH, NX, NX16, NNODES * 256);
  }

  // ---- ragged gather + concat + output projection ----
  build_cat_kernel<<<EW((int)(SB * 768)), 256, 0, stream>>>(Y1, NX16, node_map, len_context,
                                                            CAT, (int)(SB * 768));
  wmma_gemm_kernel<<<dim3((int)(SB / 32), 2), 128, 0, stream>>>(
      CAT, 768, WW16, 768, W_b, out_main, 512, 768);
}
